// EegFunctionColumn_14_5m128_28355374088690
// MI455X (gfx1250) — compile-verified, matrix operand order checked by reading, not run on hardware
//
#include <hip/hip_runtime.h>

// ---------------------------------------------------------------------------
// CDNA5 / gfx1250 implementation of the 6-layer recurrent WTA-LIF column.
//  - 16 batch rows per wave32 block; membrane state lives in LDS for all T=64.
//  - All charge GEMMs run on v_wmma_f32_16x16x32_f16 (spikes are exact {0,1}).
//  - Weights pre-converted once to f16, transposed [N][Kpad] in d_ws so the
//    WMMA B-operand loads are contiguous; global_prefetch_b8 issued per tile.
// ---------------------------------------------------------------------------

typedef __attribute__((ext_vector_type(16))) _Float16 v16h;
typedef __attribute__((ext_vector_type(8)))  float    v8f;

#define VTH   1.2f
#define DECAY (1.0f - 1.0f / 3.0f)   // 1 - 1/tau, tau = 3

struct Weights { const _Float16* w[17]; };

// ---- step program (22 GEMM+LIF ops per timestep), matches reference order --
__device__ const int OP_W[22]   = {0,1,2,3,4,5,6,7,8,9,10,11, 16,9,15,7,14,5,13,3,12,1};
__device__ const int OP_KP[22]  = {32,64,64,128,128,128,128,128,128,128,128,128,
                                   128,128,128,128,128,128,128,128,128,64};
__device__ const int OP_N[22]   = {64,64,128,128,128,128,128,128,128,128,128,128,
                                   128,128,128,128,128,128,128,128,64,64};
__device__ const int OP_V[22]   = {0,0,1,1,2,2,3,3,4,4,5,5, 4,4,3,3,2,2,1,1,0,0};
__device__ const int OP_SIN[22] = {2,0,0,1,1,0,0,1,1,0,0,1, 1,0,0,1,1,0,0,1,1,0};
__device__ const int OP_SOUT[22]= {0,-1,1,-1,0,-1,1,-1,0,-1,1,-1,
                                   0,-1,1,-1,0,-1,1,-1,0,-1};
__device__ const int VOFF[6]    = {0, 1024, 3072, 5120, 7168, 9216};

// f32 W[K][N] (row major) -> f16 Wt[N][Kpad] (transposed, zero K-padding)
__global__ void prep_weight_kernel(const float* __restrict__ src,
                                   _Float16* __restrict__ dst,
                                   int K, int N, int Kpad) {
  int idx = blockIdx.x * blockDim.x + threadIdx.x;
  if (idx >= N * Kpad) return;
  int n = idx / Kpad, k = idx % Kpad;
  dst[idx] = (_Float16)((k < K) ? src[k * N + n] : 0.0f);
}

// One wave computes X[16][N] = A[16][Kpad] @ Wt^T, accumulating f32 via WMMA.
// A is f16 in LDS (row stride Kpad); Bt is f16 in global, [N][Kpad].
__device__ __forceinline__ void gemm16(const _Float16* A,
                                       const _Float16* __restrict__ Bt,
                                       int Kpad, int N,
                                       float* __restrict__ xb, int lane) {
  const int col = lane & 15;   // A: row M index; B: column N index
  const int hs  = lane >> 4;   // half-wave select per CDNA5 16-bit layouts
  for (int nt = 0; nt < (N >> 4); ++nt) {
    const _Float16* brow = Bt + (size_t)(nt * 16 + col) * Kpad;
    __builtin_prefetch(brow, 0, 1);     // global_prefetch_b8
    v8f c = {};
    for (int kb = 0; kb < Kpad; kb += 32) {
      v16h a, b;
#pragma unroll
      for (int h = 0; h < 16; ++h) {
        // 16-bit A/B element layout: k = kb + (h<8 ? h : h+8) + 8*hs
        int k = kb + ((h < 8) ? h : (h + 8)) + 8 * hs;
        a[h] = A[col * Kpad + k];
        b[h] = brow[k];
      }
      c = __builtin_amdgcn_wmma_f32_16x16x32_f16(
          /*neg_a=*/false, a, /*neg_b=*/false, b,
          /*c_mod=*/(short)0, c, /*reuse_a=*/false, /*reuse_b=*/false);
    }
#pragma unroll
    for (int r = 0; r < 8; ++r)   // C/D layout: VGPR r -> M = r + 8*hs
      xb[(r + 8 * hs) * 128 + nt * 16 + col] = c[r];
  }
}

// LIF: charge (decay_input=False), per-row WTA spike, hard reset at VTH.
__device__ __forceinline__ void lif16(float* __restrict__ v, int N,
                                      const float* __restrict__ xb,
                                      _Float16* __restrict__ sOut, int lane) {
  if (lane < 16) {
    const int r = lane;
    float vmax = -3.4e38f; int amax = 0;
    for (int j = 0; j < N; ++j) {
      float vn = v[r * N + j] * DECAY + xb[r * 128 + j];
      v[r * N + j] = vn;
      if (vn > vmax) { vmax = vn; amax = j; }   // first-max (jnp.argmax)
    }
    const float fired = (vmax >= VTH) ? 1.0f : 0.0f;  // heaviside(x>=0)
    for (int j = 0; j < N; ++j) {
      float vn = v[r * N + j];
      if (sOut) sOut[r * N + j] = (_Float16)((j == amax) ? fired : 0.0f);
      v[r * N + j] = (vn >= VTH) ? 0.0f : vn;   // un-masked hard reset
    }
  }
}

__global__ __launch_bounds__(32)
void snn_column_kernel(const float* __restrict__ x, Weights W,
                       float* __restrict__ out) {
  __shared__ _Float16 sA[16 * 32];        // input slice, Kpad = 32
  __shared__ _Float16 sS[2][16 * 128];    // ping-pong one-hot spike buffers
  __shared__ float    sX[16 * 128];       // GEMM output / charge buffer
  __shared__ float    sV[11264];          // v1(16x64) + v2..v6(16x128 each)

  const int lane    = threadIdx.x;
  const int rowbase = blockIdx.x * 16;

  for (int i = lane; i < 11264;  i += 32) sV[i] = 0.0f;
  for (int i = lane; i < 16 * 32; i += 32) sA[i] = (_Float16)0.0f;
  __syncthreads();

#pragma unroll 1
  for (int t = 0; t < 64; ++t) {
    // gather x[:, :, t] for our 16 rows (strided by T=64 in memory)
    for (int i = lane; i < 16 * 14; i += 32) {
      int r = i / 14, l = i % 14;
      sA[r * 32 + l] = (_Float16)x[((size_t)(rowbase + r) * 14 + l) * 64 + t];
    }
    __syncthreads();
#pragma unroll 1
    for (int op = 0; op < 22; ++op) {
      const _Float16* Ain = (OP_SIN[op] == 2) ? sA : sS[OP_SIN[op]];
      gemm16(Ain, W.w[OP_W[op]], OP_KP[op], OP_N[op], sX, lane);
      __syncthreads();
      lif16(sV + VOFF[OP_V[op]], OP_N[op], sX,
            (OP_SOUT[op] < 0) ? (_Float16*)nullptr : sS[OP_SOUT[op]], lane);
      __syncthreads();
    }
  }

  // outputs: exp(v2), exp(v3), exp(v4), exp(v5), exp(v6), each [1024,128]
  for (int lidx = 0; lidx < 5; ++lidx) {
    const float* v = sV + VOFF[1 + lidx];
    for (int i = lane; i < 16 * 128; i += 32) {
      int r = i / 128, j = i % 128;
      out[((size_t)lidx * 1024 + rowbase + r) * 128 + j] = expf(v[r * 128 + j]);
    }
  }
}

extern "C" void kernel_launch(void* const* d_in, const int* in_sizes, int n_in,
                              void* d_out, int out_size, void* d_ws, size_t ws_size,
                              hipStream_t stream) {
  // params in sorted-name order after x:
  // b1_bridge,b1_inside,b2_bridge,b2_inside,b3_bridge,b3_inside,b4_bridge,
  // b4_inside,b5_bridge,b5_inside,b6_bridge,b6_inside,r21,r32,r43,r54,r65
  static const int KS[17] = {14,64,64,128,128,128,128,128,128,128,128,128,
                             128,128,128,128,128};
  static const int NS[17] = {64,64,128,128,128,128,128,128,128,128,128,128,
                             64,128,128,128,128};

  (void)in_sizes; (void)n_in; (void)out_size; (void)ws_size;

  const float* x = (const float*)d_in[0];
  _Float16* ws = (_Float16*)d_ws;

  Weights W;
  size_t off = 0;
  for (int i = 0; i < 17; ++i) {
    const int K = KS[i], N = NS[i];
    const int Kpad = (K + 31) & ~31;
    const int total = N * Kpad;
    _Float16* dst = ws + off;
    W.w[i] = dst;
    prep_weight_kernel<<<(total + 255) / 256, 256, 0, stream>>>(
        (const float*)d_in[1 + i], dst, K, N, Kpad);
    off += (size_t)total;   // total f16 footprint ~460 KB in d_ws
  }

  snn_column_kernel<<<64, 32, 0, stream>>>(x, W, (float*)d_out);
}
